// MSGGeneratorLSTM_12412455486035
// MI455X (gfx1250) — compile-verified, matrix operand order checked by reading, not run on hardware
//
#include <hip/hip_runtime.h>

#define HID   512
#define BATCH 16384
#define VOC   32
#define STEPS 20

#define KC     64                 // K-slab per LDS stage
#define NSLAB  (HID / KC)         // 8
#define LPITCH 72                 // LDS row pitch in f16 (144B = 36 dwords)

typedef __attribute__((ext_vector_type(16))) _Float16 v16h;
typedef __attribute__((ext_vector_type(8)))  _Float16 v8h;
typedef __attribute__((ext_vector_type(8)))  float    v8f;

union FragU { v16h v; v8h h[2]; };

// Load a 16-lane-row fragment in the CDNA5 16-bit A/B VGPR layout.
// `p` already includes the lane-half element offset (+half*8); chunks sit at
// p+0 and p+16 elements -> two immediate-offset 16B loads (global or LDS).
template <typename PtrT>
__device__ __forceinline__ v16h load_frag(PtrT p) {
    FragU u;
    u.h[0] = *(const v8h*)(p);        // K {0..7}   or {8..15}
    u.h[1] = *(const v8h*)(p + 16);   // K {16..23} or {24..31}
    return u.v;
}

// Element offset of B-tile j (j = gate*2 + ntile) inside W_hh, relative to
// the row (col0 + r): ((j>>1)*512 + (j&1)*16) rows of 512 elements.
__device__ __forceinline__ size_t bfrag_off(int j) {
    return (size_t)((j >> 1) * HID + (j & 1) * 16) * HID;
}

// Element offset of B-tile j inside the LDS slab (rows of LPITCH f16).
__device__ __forceinline__ size_t lfrag_off(int j) {
    return (size_t)((j >> 1) * 32 + (j & 1) * 16) * LPITCH;
}

// -------------------------------------------------------------------------
// Init: f16 copies of weights, fused gate bias (x_proj + b_hh), state copy.
// -------------------------------------------------------------------------
__global__ void init_kernel(const float* __restrict__ enc_h,
                            const float* __restrict__ enc_c,
                            const float* __restrict__ W_ih,
                            const float* __restrict__ b_ih,
                            const float* __restrict__ W_hh,
                            const float* __restrict__ b_hh,
                            const float* __restrict__ W_out,
                            const float* __restrict__ x0,
                            _Float16* __restrict__ h16,
                            float*    __restrict__ c,
                            _Float16* __restrict__ Whh16,
                            _Float16* __restrict__ Wout16,
                            float*    __restrict__ bias,
                            float*    __restrict__ mask)
{
    size_t i = (size_t)blockIdx.x * blockDim.x + threadIdx.x;
    const size_t NH = (size_t)BATCH * HID;
    if (i < NH) {
        h16[i] = (_Float16)enc_h[i];
        c[i]   = enc_c[i];
    }
    if (i < (size_t)4 * HID * HID) Whh16[i]  = (_Float16)W_hh[i];
    if (i < (size_t)VOC * HID)     Wout16[i] = (_Float16)W_out[i];
    if (i < 4 * HID) {
        float s = b_ih[i] + b_hh[i];                 // x_proj bias part
        for (int v = 0; v < VOC; ++v)                // + W_ih @ init_input
            s += W_ih[i * VOC + v] * x0[v];
        bias[i] = s;
    }
    if (i < BATCH) mask[i] = 1.0f;
}

// -------------------------------------------------------------------------
// One LSTM step: gates = h @ W_hh.T + bias, then fused cell update.
// Block = 64 batch rows x 32 hidden cols; 4 waves, each wave 16 rows.
// W_hh is staged through double-buffered LDS in 8 K-slabs of 64:
//   slab s+1 global loads (8 x b128/thread) overlap the 16 WMMAs of slab s,
//   whose B operands come from conflict-free ds_load_b128 (pitch 72 f16).
// -------------------------------------------------------------------------
__global__ void __launch_bounds__(128)
lstm_step_kernel(const _Float16* __restrict__ h_in,
                 _Float16*       __restrict__ h_out,
                 float*          __restrict__ c,
                 const _Float16* __restrict__ Whh,
                 const float*    __restrict__ bias)
{
    __shared__ _Float16 lds_w[2][128 * LPITCH];    // 2 x 18KB

    const int tid  = threadIdx.x;
    const int lane = tid & 31;
    const int wid  = tid >> 5;
    const int half = lane >> 4;
    const int m    = lane & 15;
    const int row0 = blockIdx.x * 64 + wid * 16;   // this wave's 16 rows
    const int col0 = blockIdx.y * 32;              // block's 32 hidden cols

    v8f acc[8];                                    // acc[g*2 + t]
    #pragma unroll
    for (int j = 0; j < 8; ++j)
        #pragma unroll
        for (int e = 0; e < 8; ++e) acc[j][e] = 0.0f;

    // ---- cooperative copy mapping: thread -> (LDS row r0, 16B chunk ci) ----
    const int r0 = tid >> 3;                       // 0..15
    const int ci = tid & 7;                        // 0..7 (16B chunks per row)
    // chunk j covers LDS row r0 + 16*j  <->  W_hh row (j>>1)*512+col0+(j&1)*16+r0
    const _Float16* gsrc = Whh + (size_t)(col0 + r0) * HID + ci * 8;
    _Float16* ldst0 = &lds_w[0][(size_t)r0 * LPITCH + ci * 8];
    _Float16* ldst1 = &lds_w[1][(size_t)r0 * LPITCH + ci * 8];

    // ---- per-wave bases for A (global) and B (LDS) fragments ----
    const _Float16* hbase = h_in + (size_t)(row0 + m) * HID + half * 8;
    const _Float16* lbase0 = &lds_w[0][(size_t)m * LPITCH + half * 8];
    const _Float16* lbase1 = &lds_w[1][(size_t)m * LPITCH + half * 8];

    // Prologue: slab 0 -> LDS buffer 0.
    v8h stage[8];
    #pragma unroll
    for (int j = 0; j < 8; ++j)
        stage[j] = *(const v8h*)(gsrc + bfrag_off(j));
    v16h a0 = load_frag(hbase);
    v16h a1 = load_frag(hbase + 32);
    #pragma unroll
    for (int j = 0; j < 8; ++j)
        *(v8h*)(ldst0 + (size_t)j * 16 * LPITCH) = stage[j];
    __syncthreads();

    #pragma unroll
    for (int s = 0; s < NSLAB; ++s) {
        const int cb = s & 1;
        const _Float16* lbase = cb ? lbase1 : lbase0;
        _Float16*       nldst = cb ? ldst0  : ldst1;

        // Issue next slab's global loads (overlap with WMMAs below).
        v8h  nstage[8];
        v16h a0n, a1n;
        if (s + 1 < NSLAB) {                       // compile-time resolved
            const int kb = (s + 1) * KC;
            #pragma unroll
            for (int j = 0; j < 8; ++j)
                nstage[j] = *(const v8h*)(gsrc + bfrag_off(j) + kb);
            a0n = load_frag(hbase + (s + 1) * KC);
            a1n = load_frag(hbase + (s + 1) * KC + 32);
        }

        // Compute this slab: 2 K-steps x 8 WMMAs, B from LDS.
        #pragma unroll
        for (int ks = 0; ks < 2; ++ks) {
            v16h a = ks ? a1 : a0;
            #pragma unroll
            for (int j = 0; j < 8; ++j) {
                v16h b = load_frag(lbase + lfrag_off(j) + ks * 32);
                acc[j] = __builtin_amdgcn_wmma_f32_16x16x32_f16(
                    false, a, false, b, (short)0, acc[j], false, false);
            }
        }
        __syncthreads();                           // done reading other buffer

        if (s + 1 < NSLAB) {
            #pragma unroll
            for (int j = 0; j < 8; ++j)
                *(v8h*)(nldst + (size_t)j * 16 * LPITCH) = nstage[j];
            a0 = a0n;
            a1 = a1n;
        }
        __syncthreads();                           // slab s+1 visible
    }

    // Epilogue: D[v] holds element (row = v + 8*half, col = n = m) of tile.
    const float*    bbase  = bias  + col0 + m;
    float*          cbase  = c     + (size_t)(row0 + 8 * half) * HID + col0 + m;
    _Float16*       hobase = h_out + (size_t)(row0 + 8 * half) * HID + col0 + m;

    #pragma unroll
    for (int t = 0; t < 2; ++t) {
        #pragma unroll
        for (int v = 0; v < 8; ++v) {
            float gi = acc[0 * 2 + t][v] + bbase[0 * HID + t * 16];
            float gf = acc[1 * 2 + t][v] + bbase[1 * HID + t * 16];
            float gg = acc[2 * 2 + t][v] + bbase[2 * HID + t * 16];
            float go = acc[3 * 2 + t][v] + bbase[3 * HID + t * 16];
            float i_ = 1.0f / (1.0f + __expf(-gi));
            float f_ = 1.0f / (1.0f + __expf(-gf));
            float g_ = tanhf(gg);
            float o_ = 1.0f / (1.0f + __expf(-go));
            size_t off = (size_t)v * HID + t * 16;
            float cn = f_ * cbase[off] + i_ * g_;
            cbase[off]  = cn;
            hobase[off] = (_Float16)(o_ * tanhf(cn));
        }
    }
}

// -------------------------------------------------------------------------
// Vocab projection + softmax + argmax + one-hot + mask + lp partials.
// One wave32 per batch row; lane = vocab index (V == 32 == wave32).
// -------------------------------------------------------------------------
__global__ void __launch_bounds__(256)
vocab_kernel(const _Float16* __restrict__ h,
             const _Float16* __restrict__ Wout,
             const float*    __restrict__ b_out,
             float*          __restrict__ mask,
             float*          __restrict__ msg_out,
             float*          __restrict__ mask_out,
             float*          __restrict__ partials,
             const int*      __restrict__ eos_ptr,
             int step)
{
    __shared__ float sred[8];
    const int lane = threadIdx.x & 31;
    const int wid  = threadIdx.x >> 5;
    const int row  = blockIdx.x * 8 + wid;
    const int eos  = eos_ptr[0];

    const _Float16* hrow = h    + (size_t)row  * HID;
    const _Float16* wrow = Wout + (size_t)lane * HID;
    float acc = 0.0f;
    #pragma unroll
    for (int k = 0; k < HID; k += 8) {
        v8h hv = *(const v8h*)(hrow + k);
        v8h wv = *(const v8h*)(wrow + k);
        #pragma unroll
        for (int j = 0; j < 8; ++j) acc += (float)hv[j] * (float)wv[j];
    }
    float logit = acc + b_out[lane];

    // wave-wide argmax (tie -> lowest index, matching jnp.argmax)
    float mval = logit;
    int   midx = lane;
    #pragma unroll
    for (int off = 16; off > 0; off >>= 1) {
        float ov = __shfl_xor(mval, off, 32);
        int   oi = __shfl_xor(midx, off, 32);
        if (ov > mval || (ov == mval && oi < midx)) { mval = ov; midx = oi; }
    }
    // sum of exp(logit - max); prob at argmax = 1/sum
    float se = __expf(logit - mval);
    #pragma unroll
    for (int off = 16; off > 0; off >>= 1) se += __shfl_xor(se, off, 32);

    msg_out[((size_t)step * BATCH + row) * VOC + lane] =
        (lane == midx) ? 1.0f : 0.0f;

    float old_mask = mask[row];
    if (lane == 0) {
        mask_out[(size_t)step * BATCH + row] = old_mask;         // pre-update
        mask[row] = (midx == eos) ? 0.0f : old_mask;
        sred[wid] = -__logf(se) * old_mask;                      // log p_max
    }
    __syncthreads();
    if (threadIdx.x == 0) {
        float s = 0.0f;
        #pragma unroll
        for (int w = 0; w < 8; ++w) s += sred[w];
        partials[blockIdx.x] = s;
    }
}

// Deterministic tree reduction of lp partials; accumulates into d_out scalar.
__global__ void __launch_bounds__(256)
reduce_kernel(const float* __restrict__ partials, float* __restrict__ lp,
              int step, int n)
{
    __shared__ float s[256];
    float a = 0.0f;
    for (int i = threadIdx.x; i < n; i += 256) a += partials[i];
    s[threadIdx.x] = a;
    __syncthreads();
    for (int off = 128; off > 0; off >>= 1) {
        if (threadIdx.x < off) s[threadIdx.x] += s[threadIdx.x + off];
        __syncthreads();
    }
    if (threadIdx.x == 0) {
        if (step == 0) lp[0] = s[0];
        else           lp[0] += s[0];
    }
}

// -------------------------------------------------------------------------
extern "C" void kernel_launch(void* const* d_in, const int* in_sizes, int n_in,
                              void* d_out, int out_size, void* d_ws, size_t ws_size,
                              hipStream_t stream)
{
    const float* enc_h = (const float*)d_in[0];
    const float* enc_c = (const float*)d_in[1];
    const float* W_ih  = (const float*)d_in[2];
    const float* b_ih  = (const float*)d_in[3];
    const float* W_hh  = (const float*)d_in[4];
    const float* b_hh  = (const float*)d_in[5];
    const float* W_out = (const float*)d_in[6];
    const float* b_out = (const float*)d_in[7];
    const float* x0    = (const float*)d_in[8];
    const int*   eos   = (const int*)d_in[9];

    char* ws = (char*)d_ws;
    auto carve = [&](size_t bytes) -> char* {
        char* p = ws;
        ws += (bytes + 255) & ~(size_t)255;
        return p;
    };
    _Float16* Whh16  = (_Float16*)carve((size_t)4 * HID * HID * 2);
    _Float16* Wout16 = (_Float16*)carve((size_t)VOC * HID * 2);
    float*    bias   = (float*)   carve((size_t)4 * HID * 4);
    _Float16* h16a   = (_Float16*)carve((size_t)BATCH * HID * 2);
    _Float16* h16b   = (_Float16*)carve((size_t)BATCH * HID * 2);
    float*    c      = (float*)   carve((size_t)BATCH * HID * 4);
    float*    mask   = (float*)   carve((size_t)BATCH * 4);
    float*    parts  = (float*)   carve((size_t)(BATCH / 8) * 4);

    float* msg      = (float*)d_out;                           // [20,B,V]
    float* mask_out = msg + (size_t)STEPS * BATCH * VOC;       // [20,1,B]
    float* lp       = mask_out + (size_t)STEPS * BATCH;        // scalar

    {
        const size_t NH = (size_t)BATCH * HID;
        int blocks = (int)((NH + 255) / 256);
        init_kernel<<<blocks, 256, 0, stream>>>(
            enc_h, enc_c, W_ih, b_ih, W_hh, b_hh, W_out, x0,
            h16a, c, Whh16, Wout16, bias, mask);
    }

    _Float16* hbuf[2] = { h16a, h16b };
    for (int t = 0; t < STEPS; ++t) {
        const _Float16* hin  = hbuf[t & 1];
        _Float16*       hout = hbuf[(t + 1) & 1];

        dim3 g1(BATCH / 64, HID / 32);
        lstm_step_kernel<<<g1, 128, 0, stream>>>(hin, hout, c, Whh16, bias);

        vocab_kernel<<<BATCH / 8, 256, 0, stream>>>(
            hout, Wout16, b_out, mask, msg, mask_out, parts, eos, t);

        reduce_kernel<<<1, 256, 0, stream>>>(parts, lp, t, BATCH / 8);
    }
}
